// KimiDeltaAttention_81570018885806
// MI455X (gfx1250) — matrix operand order repaired and checked by reading, not compiled, not run on hardware
//
#include <hip/hip_runtime.h>
#include <hip/hip_bf16.h>

// ---------------------------------------------------------------------------
// KimiDeltaAttention for MI455X (gfx1250, wave32, WMMA)
//   B=2 T=2048 HID=2048 H=16 DK=DV=128 KC=4
// ---------------------------------------------------------------------------

#define Bc   2
#define Tc   2048
#define HIDc 2048
#define Hc   16
#define DKc  128
#define DVc  128
#define KCc  4
#define MTc  (Bc * Tc)          // 4096 rows of the token matrix

typedef __attribute__((ext_vector_type(16))) __bf16 v16bf;
typedef __attribute__((ext_vector_type(8)))  float  v8f;

union Frag16 { unsigned short u[16]; uint4 q[2]; v16bf v; };

static __device__ __forceinline__ unsigned short kda_f32_to_bf16(float f) {
    unsigned int x = __float_as_uint(f);
    unsigned int r = (x + 0x7fffu + ((x >> 16) & 1u)) >> 16;   // RNE
    return (unsigned short)r;
}

// ------------------------------- f32 -> bf16 -------------------------------
__global__ void kda_cvt_bf16(const float* __restrict__ in,
                             unsigned short* __restrict__ out, int n) {
    int i = blockIdx.x * blockDim.x + threadIdx.x;
    if (i < n) out[i] = kda_f32_to_bf16(in[i]);
}

// ------------------------------- WMMA GEMM ---------------------------------
// C(f32, MxN) = A(bf16, MxK, row-major) * B(bf16, KxN, row-major)
// REQUIRES M%256==0, N%128==0, K%32==0 (all call sites satisfy this).
// 256x128 block tile, 256 threads = 8 waves in a 4x2 grid; each wave owns a
// 64x64 sub-tile = 4 A-frags x 4 B-frags -> 16 v_wmma per 32-wide K-step,
// fed by 16 ds_load_b128 (1.0 LDS load per WMMA). LDS double-buffered; B is
// staged pre-swizzled into the per-lane WMMA operand layout. Staging uses
// precomputed per-thread pointers bumped by constant strides (no per-tile
// 64-bit address rebuilds).
__global__ __launch_bounds__(256)
void kda_gemm_bf16(const unsigned short* __restrict__ A,
                   const unsigned short* __restrict__ Bm,
                   float* __restrict__ C, int M, int N, int K) {
    __shared__ __align__(16) unsigned short As[2][256 * 32];     // 16 KB each
    __shared__ __align__(16) unsigned short Bf[2][8 * 32 * 16];  //  8 KB each

    const int tid  = threadIdx.x;
    const int wave = tid >> 5;
    const int lane = tid & 31;
    const int half = lane >> 4;          // 0 | 1
    const int l16  = lane & 15;
    const int khalfA = half * 16;        // A frag: K 0-15 / 16-31 (contiguous)
    const int wm = wave >> 1;            // 0..3 : rows [wm*64, +64)
    const int wn = wave & 1;             // 0..1 : cols [wn*64, +64)

    const int m0 = blockIdx.y * 256;
    const int n0 = blockIdx.x * 128;
    const int nk = K >> 5;

    // ---- per-thread staging pointers/offsets (constant strides per tile) --
    // A tile 256x32 = 1024 uint4; thread handles idx = j*256+tid, j=0..3
    const unsigned short* pA[4];
    int dstA[4];
#pragma unroll
    for (int j = 0; j < 4; ++j) {
        int idx = j * 256 + tid;
        int r = idx >> 2, c8 = (idx & 3) * 8;
        pA[j]   = A + (size_t)(m0 + r) * K + c8;
        dstA[j] = r * 32 + c8;
    }
    // B tile 32x128: thread stages rows (r0, r0+1) x 8 cols; pair of bf16
    // packed into one dword LDS store (i0 even -> 4B aligned).
    const int r0 = (tid >> 4) * 2;
    const int cB = (tid & 15) * 8;
    const unsigned short* pB0 = Bm + (size_t)r0 * N + n0 + cB;
    const unsigned short* pB1 = pB0 + N;
    const size_t bStride = (size_t)32 * N;
    const int i0 = r0 & 15;              // even
    const int lh = (r0 >> 4) << 4;

    auto stage = [&](int buf) {
#pragma unroll
        for (int j = 0; j < 4; ++j) {
            uint4 av = *(const uint4*)pA[j];
            pA[j] += 32;
            *(uint4*)&As[buf][dstA[j]] = av;
        }
        union { uint4 v; unsigned short s[8]; } b0, b1;
        b0.v = *(const uint4*)pB0;
        b1.v = *(const uint4*)pB1;
        __builtin_prefetch(pB0 + bStride, 0, 0);     // speculative, next tile
        __builtin_prefetch(pA[0], 0, 0);
        pB0 += bStride;
        pB1 += bStride;
#pragma unroll
        for (int e = 0; e < 8; ++e) {
            int cc = cB + e;
            int slot = (((cc >> 4) * 32) + (lh | (cc & 15))) * 16 + i0;
            unsigned int packed = (unsigned int)b0.s[e] | ((unsigned int)b1.s[e] << 16);
            *(unsigned int*)&Bf[buf][slot] = packed;
        }
    };

    v8f zero = {0.f, 0.f, 0.f, 0.f, 0.f, 0.f, 0.f, 0.f};
    v8f acc[4][4];
#pragma unroll
    for (int at = 0; at < 4; ++at)
#pragma unroll
        for (int nt = 0; nt < 4; ++nt) acc[at][nt] = zero;

    stage(0);
    __syncthreads();

    for (int kt = 0; kt < nk; ++kt) {
        const int cur = kt & 1;
        if (kt + 1 < nk) stage(cur ^ 1);

        Frag16 b[4];
#pragma unroll
        for (int nt = 0; nt < 4; ++nt) {
            const uint4* p = (const uint4*)&Bf[cur][((wn * 4 + nt) * 32 + lane) * 16];
            b[nt].q[0] = p[0];
            b[nt].q[1] = p[1];
        }
#pragma unroll
        for (int at = 0; at < 4; ++at) {
            Frag16 a;
            int ar = (wm * 64 + at * 16 + l16) * 32;
            a.q[0] = *(const uint4*)&As[cur][ar + khalfA];
            a.q[1] = *(const uint4*)&As[cur][ar + 8 + khalfA];
#pragma unroll
            for (int nt = 0; nt < 4; ++nt)
                acc[at][nt] = __builtin_amdgcn_wmma_f32_16x16x32_bf16(
                    false, a.v, false, b[nt].v, (short)0, acc[at][nt], false, false);
        }
        __syncthreads();
    }

    // C store: VGPR r -> row (half*8 + r), col = lane&15  (dims are exact)
#pragma unroll
    for (int at = 0; at < 4; ++at)
#pragma unroll
        for (int nt = 0; nt < 4; ++nt) {
            int growb = m0 + wm * 64 + at * 16 + half * 8;
            int gcol  = n0 + wn * 64 + nt * 16 + l16;
#pragma unroll
            for (int r = 0; r < 8; ++r)
                C[(size_t)(growb + r) * N + gcol] = acc[at][nt][r];
        }
}

// ---------- fused causal depthwise conv (KC=4) + SiLU + opt. L2-norm -------
// one 128-thread block per (token-row, head): conv+silu in registers, then
// row L2-norm (as in the reference) without extra global round-trips.
__global__ __launch_bounds__(128)
void kda_conv_silu_norm(const float* __restrict__ raw,
                        const float* __restrict__ w,     // (2048, 4)
                        float* __restrict__ out,
                        float scale, int do_norm) {
    __shared__ float red[128];
    const int row = blockIdx.x >> 4;          // b*T + t
    const int h   = blockIdx.x & 15;
    const int tid = threadIdx.x;              // 128
    const int c   = h * DKc + tid;
    const int t   = row & (Tc - 1);

    float acc = 0.f;
#pragma unroll
    for (int i = 0; i < KCc; ++i) {
        int tt = t - (KCc - 1) + i;
        if (tt >= 0) acc += raw[(size_t)(row - (KCc - 1) + i) * HIDc + c] * w[c * KCc + i];
    }
    float x = acc / (1.f + __expf(-acc));     // silu

    if (do_norm) {
        red[tid] = x * x;
        __syncthreads();
        for (int s = 64; s > 0; s >>= 1) {
            if (tid < s) red[tid] += red[tid + s];
            __syncthreads();
        }
        x = x * rsqrtf(red[0] + 1e-6f) * scale;
    }
    out[(size_t)row * HIDc + c] = x;
}

// --------------------------- KDA fused gate --------------------------------
__global__ void kda_gate(float* __restrict__ g,               // in-place on graw
                         const float* __restrict__ dt_bias,   // (2048,)
                         const float* __restrict__ A_log) {   // (16,)
    int idx = blockIdx.x * blockDim.x + threadIdx.x;
    if (idx >= MTc * HIDc) return;
    int col = idx & 2047;
    int h   = col >> 7;
    float x  = g[idx] + dt_bias[col];
    float sp = (x > 20.f) ? x : log1pf(__expf(x));
    g[idx] = __expf(-sp * __expf(A_log[h]));
}

// ------------------------------- beta --------------------------------------
__global__ void kda_beta(const float* __restrict__ X,   // (4096,2048)
                         const float* __restrict__ Wb,  // (2048,16)
                         float* __restrict__ beta) {    // (4096,16)
    int idx = blockIdx.x * blockDim.x + threadIdx.x;
    if (idx >= MTc * Hc) return;
    int row = idx >> 4, h = idx & 15;
    float acc = 0.f;
#pragma unroll 8
    for (int kk = 0; kk < HIDc; ++kk) acc += X[(size_t)row * HIDc + kk] * Wb[kk * Hc + h];
    beta[idx] = 1.f / (1.f + __expf(-acc));
}

// --------------------- gated delta-rule recurrence -------------------------
// one workgroup per (b,h); S (128x128 f32) resident in LDS; scan over T.
__global__ __launch_bounds__(256)
void kda_recurrence(const float* __restrict__ qp, const float* __restrict__ kp,
                    const float* __restrict__ vp, const float* __restrict__ gp,
                    const float* __restrict__ bp, float* __restrict__ op) {
    __shared__ float S[DKc * DVc];        // 64 KB
    __shared__ float kq[DKc], qq[DKc], vv[DVc], gg[DKc];
    __shared__ float pred2[2 * DVc];
    __shared__ float delta[DVc];

    const int b = blockIdx.x >> 4;
    const int h = blockIdx.x & 15;
    const int tid  = threadIdx.x;         // 256
    const int vcol = tid & 127;
    const int kgrp = tid >> 7;            // 0 | 1
    const int k0   = kgrp * 64;

    for (int i = tid; i < DKc * DVc; i += 256) S[i] = 0.f;
    __syncthreads();

    for (int t = 0; t < Tc; ++t) {
        size_t base = ((size_t)b * Tc + t) * HIDc + h * DKc;
        if (tid < 128) {
            kq[tid] = kp[base + tid];
            qq[tid] = qp[base + tid];
            vv[tid] = vp[base + tid];
            gg[tid] = gp[base + tid];
        }
        __syncthreads();
        float bt = bp[((size_t)b * Tc + t) * Hc + h];

        // decay S along the key dim, partial k^T S
        float p = 0.f;
#pragma unroll 8
        for (int kk = k0; kk < k0 + 64; ++kk) {
            float s = S[kk * DVc + vcol] * gg[kk];
            S[kk * DVc + vcol] = s;
            p += kq[kk] * s;
        }
        pred2[kgrp * DVc + vcol] = p;
        __syncthreads();
        if (tid < 128) {
            float pr = pred2[vcol] + pred2[DVc + vcol];
            delta[vcol] = (vv[vcol] - pr) * bt;
        }
        __syncthreads();

        // rank-1 update + partial q^T S
        float d  = delta[vcol];
        float o  = 0.f;
#pragma unroll 8
        for (int kk = k0; kk < k0 + 64; ++kk) {
            float s = S[kk * DVc + vcol] + kq[kk] * d;
            S[kk * DVc + vcol] = s;
            o += qq[kk] * s;
        }
        pred2[kgrp * DVc + vcol] = o;
        __syncthreads();
        if (tid < 128) op[base + vcol] = pred2[vcol] + pred2[DVc + vcol];
        __syncthreads();
    }
}

// ------------------- gated RMSNorm -> bf16 for final GEMM ------------------
__global__ void kda_rmsnorm_gate(const float* __restrict__ o,
                                 const float* __restrict__ g2,
                                 const float* __restrict__ norm_w,
                                 unsigned short* __restrict__ obf) {
    __shared__ float red[128];
    int row = blockIdx.x;                 // (b,t,h)
    int tid = threadIdx.x;                // 128
    size_t base = (size_t)row * DVc;
    float x = o[base + tid];
    red[tid] = x * x;
    __syncthreads();
    for (int s = 64; s > 0; s >>= 1) {
        if (tid < s) red[tid] += red[tid + s];
        __syncthreads();
    }
    float r = rsqrtf(red[0] * (1.f / DVc) + 1e-6f);
    float gate = 1.f / (1.f + __expf(-g2[base + tid]));
    obf[base + tid] = kda_f32_to_bf16(x * r * norm_w[tid] * gate);
}

// ---------------------------------------------------------------------------
extern "C" void kernel_launch(void* const* d_in, const int* in_sizes, int n_in,
                              void* d_out, int out_size, void* d_ws, size_t ws_size,
                              hipStream_t stream) {
    (void)in_sizes; (void)n_in; (void)out_size; (void)ws_size;

    const float* X      = (const float*)d_in[0];
    const float* Wq     = (const float*)d_in[1];
    const float* Wk     = (const float*)d_in[2];
    const float* Wv     = (const float*)d_in[3];
    const float* conv_q = (const float*)d_in[4];
    const float* conv_k = (const float*)d_in[5];
    const float* conv_v = (const float*)d_in[6];
    const float* A_log  = (const float*)d_in[7];
    const float* Wfa    = (const float*)d_in[8];
    const float* Wfb    = (const float*)d_in[9];
    const float* dt_b   = (const float*)d_in[10];
    const float* Wb     = (const float*)d_in[11];
    const float* Wga    = (const float*)d_in[12];
    const float* Wgb    = (const float*)d_in[13];
    const float* norm_w = (const float*)d_in[14];
    const float* Wo     = (const float*)d_in[15];

    char*  ws  = (char*)d_ws;
    size_t off = 0;
    auto take = [&](size_t bytes) -> char* {
        char* p = ws + off;
        off += (bytes + 255) & ~(size_t)255;
        return p;
    };

    const size_t M = MTc, N = HIDc, KK = HIDc;
    unsigned short* XBF   = (unsigned short*)take(M * KK * 2);
    unsigned short* WQBF  = (unsigned short*)take(KK * N * 2);
    unsigned short* WKBF  = (unsigned short*)take(KK * N * 2);
    unsigned short* WVBF  = (unsigned short*)take(KK * N * 2);
    unsigned short* WOBF  = (unsigned short*)take(KK * N * 2);
    unsigned short* WFABF = (unsigned short*)take(KK * DVc * 2);
    unsigned short* WFBBF = (unsigned short*)take(DVc * N * 2);
    unsigned short* WGABF = (unsigned short*)take(KK * DVc * 2);
    unsigned short* WGBBF = (unsigned short*)take(DVc * N * 2);
    float* QRAW = (float*)take(M * N * 4);
    float* KRAW = (float*)take(M * N * 4);
    float* VRAW = (float*)take(M * N * 4);
    float* QC   = (float*)take(M * N * 4);
    float* KCb  = (float*)take(M * N * 4);
    float* VC   = (float*)take(M * N * 4);
    float* FA   = (float*)take(M * DVc * 4);
    unsigned short* FABF = (unsigned short*)take(M * DVc * 2);
    float* GBUF = (float*)take(M * N * 4);
    float* GA   = (float*)take(M * DVc * 4);
    unsigned short* GABF = (unsigned short*)take(M * DVc * 2);
    float* G2   = (float*)take(M * N * 4);
    float* BETA = (float*)take(M * Hc * 4);
    float* OBUF = (float*)take(M * N * 4);
    unsigned short* OBF = (unsigned short*)take(M * N * 2);

    auto cvt = [&](const float* src, unsigned short* dst, size_t n) {
        kda_cvt_bf16<<<(unsigned)((n + 255) / 256), 256, 0, stream>>>(src, dst, (int)n);
    };
    auto gemm = [&](const unsigned short* A, const unsigned short* Bm, float* C,
                    int m, int n, int k) {
        dim3 grid((n + 127) / 128, (m + 255) / 256);
        kda_gemm_bf16<<<grid, 256, 0, stream>>>(A, Bm, C, m, n, k);
    };

    // 1) precision conversion
    cvt(X,   XBF,   M * KK);
    cvt(Wq,  WQBF,  KK * N);
    cvt(Wk,  WKBF,  KK * N);
    cvt(Wv,  WVBF,  KK * N);
    cvt(Wo,  WOBF,  KK * N);
    cvt(Wfa, WFABF, KK * DVc);
    cvt(Wfb, WFBBF, DVc * N);
    cvt(Wga, WGABF, KK * DVc);
    cvt(Wgb, WGBBF, DVc * N);

    // 2) q/k/v projections (WMMA)
    gemm(XBF, WQBF, QRAW, MTc, HIDc, HIDc);
    gemm(XBF, WKBF, KRAW, MTc, HIDc, HIDc);
    gemm(XBF, WVBF, VRAW, MTc, HIDc, HIDc);

    // 3) fused causal conv + SiLU (+ L2-norm for q,k; q also scaled DK^-1/2)
    kda_conv_silu_norm<<<MTc * Hc, 128, 0, stream>>>(QRAW, conv_q, QC,
                                                     0.08838834764831845f, 1);
    kda_conv_silu_norm<<<MTc * Hc, 128, 0, stream>>>(KRAW, conv_k, KCb, 1.0f, 1);
    kda_conv_silu_norm<<<MTc * Hc, 128, 0, stream>>>(VRAW, conv_v, VC, 1.0f, 0);

    // 4) low-rank gate g = exp(-softplus(X Wfa Wfb + b) * exp(A_log))
    gemm(XBF, WFABF, FA, MTc, DVc, HIDc);
    cvt(FA, FABF, M * DVc);
    gemm(FABF, WFBBF, GBUF, MTc, HIDc, DVc);
    unsigned nel = MTc * HIDc;
    kda_gate<<<nel / 256, 256, 0, stream>>>(GBUF, dt_b, A_log);

    // 5) low-rank output gate g2 = X Wga Wgb
    gemm(XBF, WGABF, GA, MTc, DVc, HIDc);
    cvt(GA, GABF, M * DVc);
    gemm(GABF, WGBBF, G2, MTc, HIDc, DVc);

    // 6) beta = sigmoid(X Wb)
    kda_beta<<<(MTc * Hc + 255) / 256, 256, 0, stream>>>(X, Wb, BETA);

    // 7) gated delta-rule scan (LDS-resident state)
    kda_recurrence<<<Bc * Hc, 256, 0, stream>>>(QC, KCb, VC, GBUF, BETA, OBUF);

    // 8) gated RMSNorm, emit bf16 for final projection
    kda_rmsnorm_gate<<<MTc * Hc, 128, 0, stream>>>(OBUF, G2, norm_w, OBF);

    // 9) output projection (WMMA) straight into d_out (f32)
    gemm(OBF, WOBF, (float*)d_out, MTc, HIDc, HIDc);
}
